// Decoder_88261577933428
// MI455X (gfx1250) — compile-verified, matrix operand order checked by reading, not run on hardware
//
#include <hip/hip_runtime.h>
#include <hip/hip_bf16.h>

#define NMESH 2562
#define NGRID 16384
#define NEDGE 49152
#define DDIM  512
#define ODIM  128

typedef __attribute__((ext_vector_type(16))) __bf16 v16bf;
typedef __attribute__((ext_vector_type(8)))  float  v8f;

__device__ __forceinline__ unsigned short f2bf(float f) {
  unsigned u = __float_as_uint(f);
  unsigned r = 0x7FFFu + ((u >> 16) & 1u);
  return (unsigned short)((u + r) >> 16);
}

union FragBF { v16bf v; uint4 q[2]; };

enum { A_BF16 = 0, A_CAT3 = 1, A_CAT2 = 2 };
enum { E_RELU_BF16 = 0, E_LN_BF16 = 1, E_LN_SCATTER = 2, E_LN_RESID = 3, E_LN_F32 = 4 };

// Fused GEMM (+bias, +ReLU or +LayerNorm epilogue) using v_wmma_f32_16x16x32_bf16.
// Workgroup: 256 threads (8 wave32), tile = 32 rows x N cols (full row for LN).
// A k-slab staged in LDS. For dense bf16 A the fill uses the CDNA5 async path
// (global_load_async_to_lds_b64, ASYNCcnt); gather modes convert fp32->bf16 in VALU.
// B = bf16 W^T streamed from L2.
template<int K, int N, int AMODE, int EPI>
__global__ __launch_bounds__(256) void gemm2l(
    const unsigned short* __restrict__ Abf,   // dense bf16 A, or third concat source (emb)
    const float* __restrict__ srcX,           // grid nodes (x_i / cat2 first half / unused)
    const float* __restrict__ srcY,           // mesh nodes (x_j) or aggr (cat2 second half)
    const int*   __restrict__ eidx,           // [2][NEDGE]: senders, receivers
    const unsigned short* __restrict__ Wt,    // bf16 W^T, [N][K]
    const float* __restrict__ bias,           // [N]
    const float* __restrict__ gW,             // LN gamma [N]
    const float* __restrict__ betaW,          // LN beta  [N]
    const float* __restrict__ resid,          // residual source (grid) for E_LN_RESID
    unsigned short* __restrict__ outbf,       // bf16 out [M][N]
    float* __restrict__ outf,                 // fp32 out [M][N]
    float* __restrict__ aggr)                 // scatter target [NGRID][DDIM]
{
  constexpr int NCT = N / 128;              // 16-wide col tiles per wave (8 waves)
  const int row0 = blockIdx.x * 32;
  const int tid  = threadIdx.x;
  const int w    = tid >> 5;
  const int lane = tid & 31;
  const int hf   = lane >> 4;               // half-wave select (K packing / M+8 rows)
  const int lr   = lane & 15;
  const int ncol0 = w * (N / 8);

  __shared__ unsigned short Alds[32][40];   // 32x32 bf16 slab, stride 40 (16B aligned, no bank conflicts)
  __shared__ float redS[8][32];
  __shared__ float redS2[8][32];
  __shared__ float rowMu[32];
  __shared__ float rowRs[32];

  // A-fill assignment: thread -> fixed (row, 4 consecutive k)
  const int arow = tid >> 3;                // 0..31
  const int akk0 = (tid & 7) * 4;           // 0,4,...,28
  int snd = 0, rcv = 0;
  if (AMODE == A_CAT3) {
    int e = row0 + arow;
    snd = eidx[e];
    rcv = eidx[NEDGE + e];
  }
  // LDS byte address of this thread's fill slot (generic ptr low 32 bits == LDS offset)
  const unsigned ldsFill = (unsigned)(uintptr_t)&Alds[arow][akk0];

  v8f acc[2][NCT] = {};

  for (int k0 = 0; k0 < K; k0 += 32) {
    __syncthreads();                         // previous slab fully consumed
    {
      int grow = row0 + arow;
      if (AMODE == A_BF16) {
        // CDNA5 async Global->LDS copy: 8 bytes (4 bf16) per lane, ASYNCcnt-tracked.
        const unsigned short* gp = Abf + (size_t)grow * K + (k0 + akk0);
        asm volatile("global_load_async_to_lds_b64 %0, %1, off"
                     :: "v"(ldsFill), "v"((unsigned long long)(uintptr_t)gp)
                     : "memory");
      } else {
        #pragma unroll
        for (int j = 0; j < 4; ++j) {
          int c = k0 + akk0 + j;
          unsigned short bv;
          if (AMODE == A_CAT3) {
            if (c < DDIM)          bv = f2bf(srcX[(size_t)rcv * DDIM + c]);
            else if (c < 2 * DDIM) bv = f2bf(srcY[(size_t)snd * DDIM + (c - DDIM)]);
            else                   bv = Abf[(size_t)grow * DDIM + (c - 2 * DDIM)];
          } else { // A_CAT2
            if (c < DDIM) bv = f2bf(srcX[(size_t)grow * DDIM + c]);
            else          bv = f2bf(srcY[(size_t)grow * DDIM + (c - DDIM)]);
          }
          Alds[arow][akk0 + j] = bv;
        }
      }
    }
    if (AMODE == A_BF16)
      asm volatile("s_wait_asynccnt 0" ::: "memory");  // own async LDS writes landed
    __syncthreads();                                    // slab published to all waves

    // A fragments (16x32 bf16): lane lr = row, halves hold K 0..7/16..23 vs 8..15/24..31
    FragBF a0, a1;
    a0.q[0] = *(const uint4*)&Alds[lr][hf * 8];
    a0.q[1] = *(const uint4*)&Alds[lr][16 + hf * 8];
    a1.q[0] = *(const uint4*)&Alds[16 + lr][hf * 8];
    a1.q[1] = *(const uint4*)&Alds[16 + lr][16 + hf * 8];

    #pragma unroll
    for (int ct = 0; ct < NCT; ++ct) {
      FragBF b;
      const unsigned short* bp = Wt + (size_t)(ncol0 + ct * 16 + lr) * K + k0;
      b.q[0] = *(const uint4*)(bp + hf * 8);
      b.q[1] = *(const uint4*)(bp + 16 + hf * 8);
      acc[0][ct] = __builtin_amdgcn_wmma_f32_16x16x32_bf16(false, a0.v, false, b.v,
                                                           (short)0, acc[0][ct], false, false);
      acc[1][ct] = __builtin_amdgcn_wmma_f32_16x16x32_bf16(false, a1.v, false, b.v,
                                                           (short)0, acc[1][ct], false, false);
    }
  }

  // bias
  #pragma unroll
  for (int ct = 0; ct < NCT; ++ct) {
    float bcol = bias[ncol0 + ct * 16 + lr];
    #pragma unroll
    for (int rt = 0; rt < 2; ++rt)
      #pragma unroll
      for (int v = 0; v < 8; ++v)
        acc[rt][ct][v] += bcol;
  }

  if (EPI == E_RELU_BF16) {
    #pragma unroll
    for (int rt = 0; rt < 2; ++rt)
      #pragma unroll
      for (int ct = 0; ct < NCT; ++ct)
        #pragma unroll
        for (int v = 0; v < 8; ++v) {
          int row = rt * 16 + v + 8 * hf;       // C layout: lanes 16-31 hold M+8
          int col = ncol0 + ct * 16 + lr;
          outbf[(size_t)(row0 + row) * N + col] = f2bf(fmaxf(acc[rt][ct][v], 0.f));
        }
    return;
  }

  // LayerNorm: per-row sum/sumsq (this wave's columns), then cross-wave via LDS
  float s[2][8], s2[2][8];
  #pragma unroll
  for (int rt = 0; rt < 2; ++rt)
    #pragma unroll
    for (int v = 0; v < 8; ++v) {
      float a = 0.f, b = 0.f;
      #pragma unroll
      for (int ct = 0; ct < NCT; ++ct) {
        float x = acc[rt][ct][v];
        a += x; b += x * x;
      }
      #pragma unroll
      for (int m = 1; m < 16; m <<= 1) {        // reduce within 16-lane half (same row)
        a += __shfl_xor(a, m, 32);
        b += __shfl_xor(b, m, 32);
      }
      s[rt][v] = a; s2[rt][v] = b;
    }
  if (lr == 0) {
    #pragma unroll
    for (int rt = 0; rt < 2; ++rt)
      #pragma unroll
      for (int v = 0; v < 8; ++v) {
        int row = rt * 16 + v + 8 * hf;
        redS[w][row]  = s[rt][v];
        redS2[w][row] = s2[rt][v];
      }
  }
  __syncthreads();
  if (tid < 32) {
    float S = 0.f, S2 = 0.f;
    #pragma unroll
    for (int ww = 0; ww < 8; ++ww) { S += redS[ww][tid]; S2 += redS2[ww][tid]; }
    float mu  = S / (float)N;
    float var = S2 / (float)N - mu * mu;
    rowMu[tid] = mu;
    rowRs[tid] = rsqrtf(var + 1e-5f);
  }
  __syncthreads();

  #pragma unroll
  for (int ct = 0; ct < NCT; ++ct) {
    int col = ncol0 + ct * 16 + lr;
    float gv = gW[col], bev = betaW[col];
    #pragma unroll
    for (int rt = 0; rt < 2; ++rt)
      #pragma unroll
      for (int v = 0; v < 8; ++v) {
        int row  = rt * 16 + v + 8 * hf;
        int grow = row0 + row;
        float val = (acc[rt][ct][v] - rowMu[row]) * rowRs[row] * gv + bev;
        if (EPI == E_LN_BF16) {
          outbf[(size_t)grow * N + col] = f2bf(val);
        } else if (EPI == E_LN_SCATTER) {
          int r2 = eidx[NEDGE + grow];
          __hip_atomic_fetch_add(&aggr[(size_t)r2 * DDIM + col], val,
                                 __ATOMIC_RELAXED, __HIP_MEMORY_SCOPE_AGENT);
        } else if (EPI == E_LN_RESID) {
          val += resid[(size_t)grow * DDIM + col];
          outbf[(size_t)grow * N + col] = f2bf(val);
        } else { // E_LN_F32
          outf[(size_t)grow * N + col] = val;
        }
      }
  }
}

// W [K][N] fp32 -> Wt [N][K] bf16 (B operand layout = same packing as A)
__global__ void wprep(const float* __restrict__ W, unsigned short* __restrict__ Wt,
                      int K, int N) {
  int gid = blockIdx.x * 256 + threadIdx.x;
  if (gid >= K * N) return;
  int n = gid / K, k = gid - n * K;
  Wt[gid] = f2bf(W[(size_t)k * N + n]);
}

// edge MLP layer 1: K=4, pure VALU: t1 = relu(attr @ w1 + b1) -> bf16
__global__ void edge_l1(const float* __restrict__ attr, const float* __restrict__ w1,
                        const float* __restrict__ b1, unsigned short* __restrict__ out) {
  int gid = blockIdx.x * 256 + threadIdx.x;   // NEDGE*512 threads exactly
  int e = gid >> 9, c = gid & 511;
  float h = b1[c];
  h += attr[e * 4 + 0] * w1[c];
  h += attr[e * 4 + 1] * w1[512 + c];
  h += attr[e * 4 + 2] * w1[1024 + c];
  h += attr[e * 4 + 3] * w1[1536 + c];
  out[gid] = f2bf(fmaxf(h, 0.f));
}

__global__ void zerof(float* __restrict__ p, int n) {
  int g = blockIdx.x * 256 + threadIdx.x;
  if (g < n) p[g] = 0.f;
}

extern "C" void kernel_launch(void* const* d_in, const int* in_sizes, int n_in,
                              void* d_out, int out_size, void* d_ws, size_t ws_size,
                              hipStream_t stream) {
  (void)in_sizes; (void)n_in; (void)out_size; (void)ws_size;
  const float* mesh    = (const float*)d_in[0];
  const float* grid    = (const float*)d_in[1];
  const float* eattr   = (const float*)d_in[2];
  const int*   eidx    = (const int*)d_in[3];
  const float* emlp_w1 = (const float*)d_in[4];
  const float* emlp_b1 = (const float*)d_in[5];
  const float* emlp_w2 = (const float*)d_in[6];
  const float* emlp_b2 = (const float*)d_in[7];
  const float* emlp_g  = (const float*)d_in[8];
  const float* emlp_be = (const float*)d_in[9];
  const float* ge_w1   = (const float*)d_in[10];
  const float* ge_b1   = (const float*)d_in[11];
  const float* ge_w2   = (const float*)d_in[12];
  const float* ge_b2   = (const float*)d_in[13];
  const float* ge_g    = (const float*)d_in[14];
  const float* ge_be   = (const float*)d_in[15];
  const float* gn_w1   = (const float*)d_in[16];
  const float* gn_b1   = (const float*)d_in[17];
  const float* gn_w2   = (const float*)d_in[18];
  const float* gn_b2   = (const float*)d_in[19];
  const float* gn_g    = (const float*)d_in[20];
  const float* gn_be   = (const float*)d_in[21];
  const float* fin_w1  = (const float*)d_in[22];
  const float* fin_b1  = (const float*)d_in[23];
  const float* fin_w2  = (const float*)d_in[24];
  const float* fin_b2  = (const float*)d_in[25];
  const float* fin_g   = (const float*)d_in[26];
  const float* fin_be  = (const float*)d_in[27];

  char* ws = (char*)d_ws;
  size_t off = 0;
  auto alloc = [&](size_t bytes) -> void* {
    void* p = ws + off;
    off = (off + bytes + 255) & ~(size_t)255;
    return p;
  };
  unsigned short* wtEmlp2 = (unsigned short*)alloc((size_t)512 * 512 * 2);
  unsigned short* wtGe1   = (unsigned short*)alloc((size_t)1536 * 512 * 2);
  unsigned short* wtGe2   = (unsigned short*)alloc((size_t)512 * 512 * 2);
  unsigned short* wtGn1   = (unsigned short*)alloc((size_t)1024 * 512 * 2);
  unsigned short* wtGn2   = (unsigned short*)alloc((size_t)512 * 512 * 2);
  unsigned short* wtFin1  = (unsigned short*)alloc((size_t)512 * 512 * 2);
  unsigned short* wtFin2  = (unsigned short*)alloc((size_t)512 * 128 * 2);
  unsigned short* bufA    = (unsigned short*)alloc((size_t)NEDGE * 512 * 2);
  unsigned short* bufB    = (unsigned short*)alloc((size_t)NEDGE * 512 * 2);
  float*          aggr    = (float*)alloc((size_t)NGRID * 512 * 4);

  auto prep = [&](const float* W, unsigned short* Wt, int K, int N) {
    int total = K * N;
    wprep<<<(total + 255) / 256, 256, 0, stream>>>(W, Wt, K, N);
  };
  prep(emlp_w2, wtEmlp2, 512, 512);
  prep(ge_w1,   wtGe1,  1536, 512);
  prep(ge_w2,   wtGe2,   512, 512);
  prep(gn_w1,   wtGn1,  1024, 512);
  prep(gn_w2,   wtGn2,   512, 512);
  prep(fin_w1,  wtFin1,  512, 512);
  prep(fin_w2,  wtFin2,  512, 128);

  // edge MLP L1 (K=4, VALU) -> bufA bf16 [E,512]
  edge_l1<<<(NEDGE * 512) / 256, 256, 0, stream>>>(eattr, emlp_w1, emlp_b1, bufA);

  // edge MLP L2 + LN -> bufB (edges_emb, bf16)
  gemm2l<512, 512, A_BF16, E_LN_BF16><<<NEDGE / 32, 256, 0, stream>>>(
      bufA, nullptr, nullptr, nullptr, wtEmlp2, emlp_b2, emlp_g, emlp_be,
      nullptr, bufB, nullptr, nullptr);

  // msg MLP L1: gathered cat(x_i, x_j, emb) @ ge_w1, ReLU -> bufA
  gemm2l<1536, 512, A_CAT3, E_RELU_BF16><<<NEDGE / 32, 256, 0, stream>>>(
      bufB, grid, mesh, eidx, wtGe1, ge_b1, nullptr, nullptr,
      nullptr, bufA, nullptr, nullptr);

  // zero aggregation target
  zerof<<<(NGRID * 512) / 256, 256, 0, stream>>>(aggr, NGRID * 512);

  // msg MLP L2 + LN + scatter-add into aggr (fp32 atomics)
  gemm2l<512, 512, A_BF16, E_LN_SCATTER><<<NEDGE / 32, 256, 0, stream>>>(
      bufA, nullptr, nullptr, eidx, wtGe2, ge_b2, ge_g, ge_be,
      nullptr, nullptr, nullptr, aggr);

  // node MLP L1: cat(grid, aggr) @ gn_w1, ReLU -> bufB
  gemm2l<1024, 512, A_CAT2, E_RELU_BF16><<<NGRID / 32, 256, 0, stream>>>(
      nullptr, grid, aggr, nullptr, wtGn1, gn_b1, nullptr, nullptr,
      nullptr, bufB, nullptr, nullptr);

  // node MLP L2 + LN + residual (grid + gnn_out) -> bufA (latent, bf16)
  gemm2l<512, 512, A_BF16, E_LN_RESID><<<NGRID / 32, 256, 0, stream>>>(
      bufB, nullptr, nullptr, nullptr, wtGn2, gn_b2, gn_g, gn_be,
      grid, bufA, nullptr, nullptr);

  // final MLP L1: latent @ fin_w1, ReLU -> bufB
  gemm2l<512, 512, A_BF16, E_RELU_BF16><<<NGRID / 32, 256, 0, stream>>>(
      bufA, nullptr, nullptr, nullptr, wtFin1, fin_b1, nullptr, nullptr,
      nullptr, bufB, nullptr, nullptr);

  // final MLP L2 + LN -> fp32 output [NGRID, 128]
  gemm2l<512, 128, A_BF16, E_LN_F32><<<NGRID / 32, 256, 0, stream>>>(
      bufB, nullptr, nullptr, nullptr, wtFin2, fin_b2, fin_g, fin_be,
      nullptr, nullptr, (float*)d_out, nullptr);
}
